// GatedConvBlock_20246475833515
// MI455X (gfx1250) — matrix operand order repaired and checked by reading, not compiled, not run on hardware
//
#include <hip/hip_runtime.h>

typedef __attribute__((ext_vector_type(16))) __bf16 v16bf;
typedef __attribute__((ext_vector_type(8)))  float  v8f;

#define DIMD   2048
#define SEQ    4096
#define MTOT   16384      // B*S
#define LDSTR  40         // padded LDS row stride (elements): 80B rows, bank stride 20

// ---------------- CDNA5 async global->LDS (ASYNCcnt path) ------------------

__device__ __forceinline__ void async_load16(unsigned lds_addr, const void* gptr) {
  // per-lane: LDS[lds_addr] = MEM[gptr], 16 bytes; tracked by ASYNCcnt
  asm volatile("global_load_async_to_lds_b128 %0, %1, off"
               :: "v"(lds_addr), "v"(gptr)
               : "memory");
}

#if defined(__has_builtin) && __has_builtin(__builtin_amdgcn_s_wait_asynccnt)
#define WAIT_ASYNC(n) __builtin_amdgcn_s_wait_asynccnt(n)
#else
#define WAIT_ASYNC(n) asm volatile("s_wait_asynccnt %0" :: "n"(n) : "memory")
#endif

__device__ __forceinline__ unsigned lds_off(const void* p) {
  // generic LDS pointer: aperture in high 32 bits, LDS byte offset in low 32
  return (unsigned)(unsigned long long)(size_t)p;
}

// ---------------- fragment loaders (per CDNA5 ISA 7.12.2 layouts) ----------

// A-matrix 16x32 bf16: lane half splits K as {0-7,16-23} / {8-15,24-31}
__device__ __forceinline__ v16bf load_frag_a(const __bf16* p) {
  const int lane = threadIdx.x & 31;
  const int r    = lane & 15;
  const int half = lane >> 4;
  v16bf f;
#pragma unroll
  for (int v = 0; v < 8; ++v) {
    const int k0 = ((v >> 2) << 4) + ((v & 3) << 1) + (half << 3);
    f[2 * v]     = p[r * LDSTR + k0];
    f[2 * v + 1] = p[r * LDSTR + k0 + 1];
  }
  return f;
}

// B-matrix 32x16 bf16: lane half splits K as {0-7,16-23} / {8-15,24-31} (col n=lane)
__device__ __forceinline__ v16bf load_frag_b(const __bf16* p) {
  const int lane = threadIdx.x & 31;
  const int n    = lane & 15;
  const int half = lane >> 4;
  v16bf f;
#pragma unroll
  for (int v = 0; v < 8; ++v) {
    const int k0 = ((v >> 2) << 4) + ((v & 3) << 1) + (half << 3);
    f[2 * v]     = p[n * LDSTR + k0];
    f[2 * v + 1] = p[n * LDSTR + k0 + 1];
  }
  return f;
}

// ---------------- kernel 1: RMS norm + bf16 cast ---------------------------

__global__ void __launch_bounds__(256) rmsnorm_kernel(const float* __restrict__ x,
                                                      __bf16* __restrict__ xnb) {
  __shared__ float red[256];
  const int row   = blockIdx.x;
  const float* xr = x + (size_t)row * DIMD;
  float vals[8];
  float s = 0.f;
#pragma unroll
  for (int i = 0; i < 8; ++i) {
    float v = xr[i * 256 + threadIdx.x];
    vals[i] = v;
    s += v * v;
  }
  red[threadIdx.x] = s;
  __syncthreads();
  for (int off = 128; off > 0; off >>= 1) {
    if (threadIdx.x < off) red[threadIdx.x] += red[threadIdx.x + off];
    __syncthreads();
  }
  const float scale = rsqrtf(red[0] * (1.0f / DIMD) + 1.1920929e-07f);
  __bf16* outr = xnb + (size_t)row * DIMD;
#pragma unroll
  for (int i = 0; i < 8; ++i)
    outr[i * 256 + threadIdx.x] = (__bf16)(vals[i] * scale);
}

// ---------------- kernel 2: fp32 -> bf16 weight cast -----------------------

__global__ void cvt_bf16_kernel(const float* __restrict__ src,
                                __bf16* __restrict__ dst, long n) {
  long i = (long)blockIdx.x * blockDim.x + threadIdx.x;
  if (i < n) dst[i] = (__bf16)src[i];
}

// ---------------- tile prefetchers (async, double-buffered) ----------------

__device__ __forceinline__ void issue_tile_g1(const __bf16* xnb, const __bf16* web,
                                              __bf16* lA, __bf16* lB,
                                              int mBase, int d0, int kt, int tid) {
  // A: 128 rows x 32 cols, 16B units -> 2 async instr / wave
#pragma unroll
  for (int it = 0; it < 2; ++it) {
    const int u = tid + it * 256;
    const int row = u >> 2, seg = u & 3;
    async_load16(lds_off(&lA[row * LDSTR + seg * 8]),
                 xnb + (((size_t)(mBase + row)) << 11) + kt + seg * 8);
  }
  // B: 3 thirds x 64 rows x 32 cols -> 3 async instr / wave
#pragma unroll
  for (int it = 0; it < 3; ++it) {
    const int u = tid + it * 256;
    const int t = u >> 8, rem = u & 255;
    const int row = rem >> 2, seg = rem & 3;
    async_load16(lds_off(&lB[t * (64 * LDSTR) + row * LDSTR + seg * 8]),
                 web + (((size_t)(t * DIMD + d0 + row)) << 11) + kt + seg * 8);
  }
}

__device__ __forceinline__ void issue_tile_g2(const __bf16* ows, const __bf16* wpb,
                                              __bf16* lA, __bf16* lB,
                                              int mBase, int n0, int kt, int tid) {
#pragma unroll
  for (int it = 0; it < 2; ++it) {
    const int u = tid + it * 256;
    const int row = u >> 2, seg = u & 3;
    async_load16(lds_off(&lA[row * LDSTR + seg * 8]),
                 ows + (((size_t)(mBase + row)) << 11) + kt + seg * 8);
  }
  {
    const int row = tid >> 2, seg = tid & 3;
    async_load16(lds_off(&lB[row * LDSTR + seg * 8]),
                 wpb + (((size_t)(n0 + row)) << 11) + kt + seg * 8);
  }
}

// ---------------- kernel 3: GEMM1 + gate epilogue --------------------------

__global__ void __launch_bounds__(256) gemm1_kernel(const __bf16* __restrict__ xnb,
                                                    const __bf16* __restrict__ web,
                                                    __bf16* __restrict__ yws,
                                                    __bf16* __restrict__ cws) {
  __shared__ __align__(16) __bf16 ldsA[2][128 * LDSTR];
  __shared__ __align__(16) __bf16 ldsB[2][3 * 64 * LDSTR];

  const int tid   = threadIdx.x;
  const int wave  = tid >> 5;
  const int wm    = wave >> 1;            // 0..3
  const int wn    = wave & 1;             // 0..1
  const int mBase = blockIdx.y * 128;
  const int d0    = blockIdx.x * 64;

  const v8f vzero = {0.f, 0.f, 0.f, 0.f, 0.f, 0.f, 0.f, 0.f};
  v8f acc[3][2][2];
#pragma unroll
  for (int t = 0; t < 3; ++t)
#pragma unroll
    for (int a = 0; a < 2; ++a)
#pragma unroll
      for (int b = 0; b < 2; ++b) acc[t][a][b] = vzero;

  issue_tile_g1(xnb, web, ldsA[0], ldsB[0], mBase, d0, 0, tid);

  const int nk = DIMD / 32;               // 64 k-steps
  for (int i = 0; i < nk; ++i) {
    const int cur = i & 1;
    WAIT_ASYNC(0);                        // own tile-i loads landed
    __syncthreads();                      // all waves: tile i visible, prev readers done
    if (i + 1 < nk)
      issue_tile_g1(xnb, web, ldsA[cur ^ 1], ldsB[cur ^ 1], mBase, d0,
                    (i + 1) * 32, tid);   // flies during compute below

    v16bf aF[2], bF[3][2];
    aF[0] = load_frag_a(&ldsA[cur][(wm * 32) * LDSTR]);
    aF[1] = load_frag_a(&ldsA[cur][(wm * 32 + 16) * LDSTR]);
#pragma unroll
    for (int t = 0; t < 3; ++t)
#pragma unroll
      for (int fb = 0; fb < 2; ++fb)
        bF[t][fb] = load_frag_b(&ldsB[cur][t * (64 * LDSTR) + (wn * 32 + fb * 16) * LDSTR]);

#pragma unroll
    for (int t = 0; t < 3; ++t)
#pragma unroll
      for (int fb = 0; fb < 2; ++fb)
#pragma unroll
        for (int fa = 0; fa < 2; ++fa)
          acc[t][fa][fb] = __builtin_amdgcn_wmma_f32_16x16x32_bf16(
              false, aF[fa], false, bF[t][fb], (short)0, acc[t][fa][fb], false, false);
  }

  // epilogue: y = Bg * z, store y and Cg (bf16)
  const int lane = tid & 31, r = lane & 15, half = lane >> 4;
#pragma unroll
  for (int fa = 0; fa < 2; ++fa)
#pragma unroll
    for (int fb = 0; fb < 2; ++fb)
#pragma unroll
      for (int i = 0; i < 8; ++i) {
        const int m = mBase + wm * 32 + fa * 16 + i + 8 * half;
        const int n = d0 + wn * 32 + fb * 16 + r;
        const size_t off = (((size_t)m) << 11) + n;
        const float bg = acc[0][fa][fb][i];
        const float cg = acc[1][fa][fb][i];
        const float zz = acc[2][fa][fb][i];
        yws[off] = (__bf16)(bg * zz);
        cws[off] = (__bf16)cg;
      }
}

// ---------------- kernel 4: causal depthwise conv + gate -------------------

__global__ void __launch_bounds__(256) conv_gate_kernel(const __bf16* __restrict__ yws,
                                                        const __bf16* __restrict__ cws,
                                                        const float* __restrict__ cw,
                                                        const float* __restrict__ cb,
                                                        const int* __restrict__ dilp,
                                                        __bf16* __restrict__ ows) {
  const long idx4 = (long)blockIdx.x * blockDim.x + threadIdx.x;
  const long e0   = idx4 * 4;
  if (e0 >= (long)MTOT * DIMD) return;
  const int m   = (int)(e0 >> 11);
  const int d   = (int)(e0 & (DIMD - 1));
  const int s   = m & (SEQ - 1);
  const int dil = dilp[0];
  const long strideD = (long)dil * DIMD;
#pragma unroll
  for (int j = 0; j < 4; ++j) {
    const int dj  = d + j;
    const float y0 = (float)yws[e0 + j];
    const float y1 = (s >= dil)     ? (float)yws[e0 + j - strideD]     : 0.f;
    const float y2 = (s >= 2 * dil) ? (float)yws[e0 + j - 2 * strideD] : 0.f;
    const float conv = cw[dj * 3 + 0] * y2 + cw[dj * 3 + 1] * y1 +
                       cw[dj * 3 + 2] * y0 + cb[dj];
    const float cg = (float)cws[e0 + j];
    ows[e0 + j] = (__bf16)(cg * conv);
  }
}

// ---------------- kernel 5: GEMM2 + residual epilogue ----------------------

__global__ void __launch_bounds__(256) gemm2_kernel(const __bf16* __restrict__ ows,
                                                    const __bf16* __restrict__ wpb,
                                                    const float* __restrict__ x,
                                                    const float* __restrict__ cs,
                                                    float* __restrict__ out) {
  __shared__ __align__(16) __bf16 ldsA[2][128 * LDSTR];
  __shared__ __align__(16) __bf16 ldsB[2][64 * LDSTR];

  const int tid   = threadIdx.x;
  const int wave  = tid >> 5;
  const int wm    = wave >> 1;
  const int wn    = wave & 1;
  const int mBase = blockIdx.y * 128;
  const int n0    = blockIdx.x * 64;

  const v8f vzero = {0.f, 0.f, 0.f, 0.f, 0.f, 0.f, 0.f, 0.f};
  v8f acc[2][2];
#pragma unroll
  for (int a = 0; a < 2; ++a)
#pragma unroll
    for (int b = 0; b < 2; ++b) acc[a][b] = vzero;

  issue_tile_g2(ows, wpb, ldsA[0], ldsB[0], mBase, n0, 0, tid);

  const int nk = DIMD / 32;
  for (int i = 0; i < nk; ++i) {
    const int cur = i & 1;
    WAIT_ASYNC(0);
    __syncthreads();
    if (i + 1 < nk)
      issue_tile_g2(ows, wpb, ldsA[cur ^ 1], ldsB[cur ^ 1], mBase, n0,
                    (i + 1) * 32, tid);

    v16bf aF[2], bF[2];
    aF[0] = load_frag_a(&ldsA[cur][(wm * 32) * LDSTR]);
    aF[1] = load_frag_a(&ldsA[cur][(wm * 32 + 16) * LDSTR]);
#pragma unroll
    for (int fb = 0; fb < 2; ++fb)
      bF[fb] = load_frag_b(&ldsB[cur][(wn * 32 + fb * 16) * LDSTR]);

#pragma unroll
    for (int fb = 0; fb < 2; ++fb)
#pragma unroll
      for (int fa = 0; fa < 2; ++fa)
        acc[fa][fb] = __builtin_amdgcn_wmma_f32_16x16x32_bf16(
            false, aF[fa], false, bF[fb], (short)0, acc[fa][fb], false, false);
  }

  const int lane = tid & 31, r = lane & 15, half = lane >> 4;
#pragma unroll
  for (int fa = 0; fa < 2; ++fa)
#pragma unroll
    for (int fb = 0; fb < 2; ++fb)
#pragma unroll
      for (int i = 0; i < 8; ++i) {
        const int m = mBase + wm * 32 + fa * 16 + i + 8 * half;
        const int n = n0 + wn * 32 + fb * 16 + r;
        const size_t off = (((size_t)m) << 11) + n;
        out[off] = x[off] + cs[n] * acc[fa][fb][i];
      }
}

// ---------------- launch ---------------------------------------------------

extern "C" void kernel_launch(void* const* d_in, const int* in_sizes, int n_in,
                              void* d_out, int out_size, void* d_ws, size_t ws_size,
                              hipStream_t stream) {
  const float* x  = (const float*)d_in[0];
  const float* We = (const float*)d_in[1];   // [3D, D]
  const float* cw = (const float*)d_in[2];   // [D,1,K]
  const float* cb = (const float*)d_in[3];   // [D]
  const float* Wp = (const float*)d_in[4];   // [D, D]
  const float* cs = (const float*)d_in[5];   // [D]
  const int*  dil = (const int*)d_in[6];
  float* out = (float*)d_out;

  char* ws = (char*)d_ws;
  size_t o = 0;
  __bf16* xnb = (__bf16*)(ws + o); o += (size_t)MTOT * DIMD * 2;      // 64 MB
  __bf16* web = (__bf16*)(ws + o); o += (size_t)3 * DIMD * DIMD * 2;  // 24 MB
  __bf16* wpb = (__bf16*)(ws + o); o += (size_t)DIMD * DIMD * 2;      //  8 MB
  __bf16* yws = (__bf16*)(ws + o); o += (size_t)MTOT * DIMD * 2;      // 64 MB
  __bf16* cwsb= (__bf16*)(ws + o); o += (size_t)MTOT * DIMD * 2;      // 64 MB
  __bf16* owsb= (__bf16*)(ws + o); o += (size_t)MTOT * DIMD * 2;      // 64 MB

  // 1. RMS norm + cast
  rmsnorm_kernel<<<MTOT, 256, 0, stream>>>(x, xnb);

  // 2. weight casts
  {
    long nWe = (long)3 * DIMD * DIMD;
    cvt_bf16_kernel<<<(int)((nWe + 255) / 256), 256, 0, stream>>>(We, web, nWe);
    long nWp = (long)DIMD * DIMD;
    cvt_bf16_kernel<<<(int)((nWp + 255) / 256), 256, 0, stream>>>(Wp, wpb, nWp);
  }

  // 3. GEMM1 + gating: grid (D/64, M/128)
  gemm1_kernel<<<dim3(DIMD / 64, MTOT / 128), 256, 0, stream>>>(xnb, web, yws, cwsb);

  // 4. causal dwconv + C-gate
  {
    long total4 = (long)MTOT * DIMD / 4;
    conv_gate_kernel<<<(int)((total4 + 255) / 256), 256, 0, stream>>>(
        yws, cwsb, cw, cb, dil, owsb);
  }

  // 5. GEMM2 + residual
  gemm2_kernel<<<dim3(DIMD / 64, MTOT / 128), 256, 0, stream>>>(owsb, wpb, x, cs, out);
}
